// PriceBranch_9320079033004
// MI455X (gfx1250) — matrix operand order repaired and checked by reading, not compiled, not run on hardware
//
#include <hip/hip_runtime.h>
#include <hip/hip_bf16.h>
#include <math.h>
#include <stdint.h>

typedef __attribute__((ext_vector_type(16))) _Float16 v16h;
typedef __attribute__((ext_vector_type(8)))  _Float16 v8h;
typedef __attribute__((ext_vector_type(8)))  float    v8f;
typedef __attribute__((ext_vector_type(4)))  unsigned int v4u;
typedef __attribute__((ext_vector_type(4)))  int v4i;
typedef __attribute__((ext_vector_type(8)))  int v8i;

#define B_DIM 4
#define T_DIM 2048
#define D_DIM 256
#define H_DIM 8
#define DK_DIM 32
#define FIN 6
#define BT (B_DIM * T_DIM)

// ---------------------------------------------------------------------------
// TDM: DMA a 32-row x 256-half (16 KB) tile of a [8192,256] f16 tensor into
// LDS. Issued by one wave (EXEC ignored by TENSOR_LOAD_TO_LDS); caller must
// __syncthreads() after. D# per CDNA5 ISA ch.8:
//   group0: count=1 | lds_addr | global_addr(57b) | type=2
//   group1: data_size=1(2B), tensor_dim0=256, tensor_dim1=8192,
//           tile_dim0=256, tile_dim1=32, tensor_dim0_stride=256
//   groups 2/3: zero (2-D tile)
// ---------------------------------------------------------------------------
__device__ __forceinline__ void tdm_load_tile_32x256(const _Float16* gsrc,
                                                     void* lds_dst) {
  unsigned long long ga = (unsigned long long)(uintptr_t)gsrc;
  unsigned int lds = (unsigned int)(uintptr_t)lds_dst;  // flat[31:0] == LDS offset
  v4u g0;
  g0[0] = 1u;                                        // count=1 (valid user D#)
  g0[1] = lds;                                       // lds_addr
  g0[2] = (unsigned int)ga;                          // global_addr[31:0]
  g0[3] = (unsigned int)(ga >> 32) | (2u << 30);     // global_addr[56:32] | type=2
  v8i g1;
  g1[0] = (int)(1u << 16);                           // data_size=1 (2 bytes)
  g1[1] = (int)(256u << 16);                         // tensor_dim0[15:0]=256
  g1[2] = (int)(8192u << 16);                        // tensor_dim1[15:0]=8192
  g1[3] = (int)(256u << 16);                         // tile_dim0=256
  g1[4] = 32;                                        // tile_dim1=32, tile_dim2=0
  g1[5] = 256;                                       // tensor_dim0_stride[31:0]=256
  g1[6] = 0;
  g1[7] = 0;
  v4i g2 = {0, 0, 0, 0};
  v4i g3 = {0, 0, 0, 0};
#if __clang_major__ >= 23
  v8i g4 = {0, 0, 0, 0, 0, 0, 0, 0};
  __builtin_amdgcn_tensor_load_to_lds(g0, g1, g2, g3, g4, 0);
#else
  __builtin_amdgcn_tensor_load_to_lds(g0, g1, g2, g3, 0);
#endif
  __builtin_amdgcn_s_wait_tensorcnt(0);
}

// ---------------------------------------------------------------------------
// Weight convert + transpose: w[K,N] f32 -> wt[N,K] f16 (B-matrix friendly)
// ---------------------------------------------------------------------------
__global__ void convw_kernel(const float* __restrict__ w, _Float16* __restrict__ wt,
                             int K, int N) {
  int idx = blockIdx.x * 256 + threadIdx.x;
  if (idx >= K * N) return;
  int k = idx / N, n = idx - k * N;
  wt[(size_t)n * K + k] = (_Float16)w[(size_t)k * N + n];
}

// ---------------------------------------------------------------------------
// Stage 1: LN(6) -> fc1(6->512)+gelu (VALU) -> fc2(512->256) via WMMA,
// + residual x@proj_w+proj_b+fc2_b. 16 rows per block, 128 threads (4 waves).
// ---------------------------------------------------------------------------
__global__ void __launch_bounds__(128) stage1_kernel(
    const float* __restrict__ x, const float* __restrict__ s1_g,
    const float* __restrict__ s1_b, const float* __restrict__ fc1_w,
    const float* __restrict__ fc1_b, const _Float16* __restrict__ fc2_wT,
    const float* __restrict__ fc2_b, const float* __restrict__ proj_w,
    const float* __restrict__ proj_b, float* __restrict__ h) {
  __shared__ float xraw[16][FIN];
  __shared__ float xln[16][FIN];
  __shared__ __align__(16) _Float16 act[16][512];
  __shared__ float res[16][256];

  const size_t r0 = (size_t)blockIdx.x * 16;
  const int tid = threadIdx.x;

  if (tid < 16) {
    float v[FIN];
    float mu = 0.f;
    for (int f = 0; f < FIN; ++f) { v[f] = x[(r0 + tid) * FIN + f]; mu += v[f]; }
    mu *= (1.f / FIN);
    float var = 0.f;
    for (int f = 0; f < FIN; ++f) { float d = v[f] - mu; var += d * d; }
    var *= (1.f / FIN);
    float rs = rsqrtf(var + 1e-5f);
    for (int f = 0; f < FIN; ++f) {
      xraw[tid][f] = v[f];
      xln[tid][f] = (v[f] - mu) * rs * s1_g[f] + s1_b[f];
    }
  }
  __syncthreads();

  // fc1 + exact GELU -> f16 activations in LDS
  for (int idx = tid; idx < 16 * 512; idx += 128) {
    int row = idx >> 9, col = idx & 511;
    float s = fc1_b[col];
    for (int f = 0; f < FIN; ++f) s += xln[row][f] * fc1_w[f * 512 + col];
    float g = 0.5f * s * (1.0f + erff(s * 0.70710678118654752f));
    act[row][col] = (_Float16)g;
  }
  // residual (raw x @ proj_w + proj_b) + fc2 bias
  for (int idx = tid; idx < 16 * 256; idx += 128) {
    int row = idx >> 8, col = idx & 255;
    float s = proj_b[col] + fc2_b[col];
    for (int f = 0; f < FIN; ++f) s += xraw[row][f] * proj_w[f * 256 + col];
    res[row][col] = s;
  }
  __syncthreads();

  // fc2: C[16x16] tiles, 4 waves x 4 N-tiles, K=512 -> 16 WMMA steps each.
  const int wave = tid >> 5, lane = tid & 31;
  const int nt0 = wave * 4;
  const int m = lane & 15, colg = lane >> 4, kb = colg * 16, ncol = lane & 15;

  v8f acc[4] = {{}, {}, {}, {}};
  for (int ks = 0; ks < 16; ++ks) {
    v16h a;
    const _Float16* ap = &act[m][ks * 32 + kb];
    v8h lo = *(const v8h*)ap, hi = *(const v8h*)(ap + 8);
    for (int i = 0; i < 8; ++i) { a[i] = lo[i]; a[8 + i] = hi[i]; }
    for (int i = 0; i < 4; ++i) {
      int n = (nt0 + i) * 16 + ncol;
      const _Float16* bp = fc2_wT + (size_t)n * 512 + ks * 32 + colg * 16;
      __builtin_prefetch(bp + 32, 0, 1);
      v8h blo = *(const v8h*)bp, bhi = *(const v8h*)(bp + 8);
      v16h bf;
      for (int j = 0; j < 8; ++j) { bf[j] = blo[j]; bf[8 + j] = bhi[j]; }
      acc[i] = __builtin_amdgcn_wmma_f32_16x16x32_f16(false, a, false, bf,
                                                      (short)0, acc[i], false, false);
    }
  }
  for (int i = 0; i < 4; ++i) {
    int n = (nt0 + i) * 16 + ncol;
    for (int r = 0; r < 8; ++r) {
      int mr = r + 8 * colg;
      h[(r0 + mr) * 256 + n] = acc[i][r] + res[mr][n];
    }
  }
}

// ---------------------------------------------------------------------------
// LayerNorm over D=256 -> f16. One wave per row, 8 rows per 256-thread block.
// ---------------------------------------------------------------------------
__global__ void __launch_bounds__(256) ln_kernel(
    const float* __restrict__ h, const float* __restrict__ g,
    const float* __restrict__ b, _Float16* __restrict__ n16) {
  const int wave = threadIdx.x >> 5, lane = threadIdx.x & 31;
  const size_t row = (size_t)blockIdx.x * 8 + wave;
  const float* hp = h + row * 256 + lane * 8;
  float v[8];
  float4 a0 = *(const float4*)hp, a1 = *(const float4*)(hp + 4);
  v[0] = a0.x; v[1] = a0.y; v[2] = a0.z; v[3] = a0.w;
  v[4] = a1.x; v[5] = a1.y; v[6] = a1.z; v[7] = a1.w;
  float s1 = 0.f, s2 = 0.f;
  for (int i = 0; i < 8; ++i) { s1 += v[i]; s2 += v[i] * v[i]; }
  for (int off = 16; off >= 1; off >>= 1) {
    s1 += __shfl_xor(s1, off, 32);
    s2 += __shfl_xor(s2, off, 32);
  }
  float mu = s1 * (1.f / 256.f);
  float var = s2 * (1.f / 256.f) - mu * mu;
  float rs = rsqrtf(var + 1e-5f);
  int d0 = lane * 8;
  v8h o;
  for (int i = 0; i < 8; ++i)
    o[i] = (_Float16)((v[i] - mu) * rs * g[d0 + i] + b[d0 + i]);
  *(v8h*)(n16 + row * 256 + d0) = o;
}

// ---------------------------------------------------------------------------
// QKV projections: [8192,256] @ [256,256] -> f16, grid.y selects q/k/v.
// 32x256 tile per block (8 waves); A staged in LDS by the TDM; WMMA K=256.
// 1/sqrt(DK) folded into Q output.
// ---------------------------------------------------------------------------
__global__ void __launch_bounds__(256) qkv_kernel(
    const _Float16* __restrict__ A, const _Float16* __restrict__ wqT,
    const _Float16* __restrict__ wkT, const _Float16* __restrict__ wvT,
    _Float16* __restrict__ q, _Float16* __restrict__ k,
    _Float16* __restrict__ v) {
  __shared__ __align__(16) _Float16 As[32][256];
  const int tid = threadIdx.x;
  const size_t r0 = (size_t)blockIdx.x * 32;
  const _Float16* WT = (blockIdx.y == 0) ? wqT : (blockIdx.y == 1) ? wkT : wvT;
  _Float16* out = (blockIdx.y == 0) ? q : (blockIdx.y == 1) ? k : v;
  const float scale = (blockIdx.y == 0) ? 0.17677669529663687f : 1.0f;

  if (tid < 32) tdm_load_tile_32x256(A + r0 * 256, &As[0][0]);
  __syncthreads();

  const int wave = tid >> 5, lane = tid & 31;
  const int mt = wave >> 2, nt0 = (wave & 3) * 4;
  const int m = lane & 15, colg = lane >> 4, kb = colg * 16, ncol = lane & 15;

  v8f acc[4] = {{}, {}, {}, {}};
  for (int ks = 0; ks < 8; ++ks) {
    v16h a;
    const _Float16* ap = &As[mt * 16 + m][ks * 32 + kb];
    v8h lo = *(const v8h*)ap, hi = *(const v8h*)(ap + 8);
    for (int i = 0; i < 8; ++i) { a[i] = lo[i]; a[8 + i] = hi[i]; }
    for (int i = 0; i < 4; ++i) {
      int n = (nt0 + i) * 16 + ncol;
      const _Float16* bp = WT + (size_t)n * 256 + ks * 32 + colg * 16;
      __builtin_prefetch(bp + 32, 0, 1);
      v8h blo = *(const v8h*)bp, bhi = *(const v8h*)(bp + 8);
      v16h bf;
      for (int j = 0; j < 8; ++j) { bf[j] = blo[j]; bf[8 + j] = bhi[j]; }
      acc[i] = __builtin_amdgcn_wmma_f32_16x16x32_f16(false, a, false, bf,
                                                      (short)0, acc[i], false, false);
    }
  }
  for (int i = 0; i < 4; ++i) {
    int n = (nt0 + i) * 16 + ncol;
    for (int r = 0; r < 8; ++r) {
      size_t row = r0 + mt * 16 + r + 8 * colg;
      out[row * 256 + n] = (_Float16)(acc[i][r] * scale);
    }
  }
}

// ---------------------------------------------------------------------------
// Windowed causal attention. Block = (16 query rows) x batch; wave = head.
// DK=32 == WMMA K. Scores: 3 WMMAs (48-key window). Softmax in C-layout via
// 16-lane shfl reductions. P -> LDS -> A-layout, PV: 2x2 WMMAs.
// ---------------------------------------------------------------------------
__global__ void __launch_bounds__(256) attn_kernel(
    const _Float16* __restrict__ q16, const _Float16* __restrict__ k16,
    const _Float16* __restrict__ v16, _Float16* __restrict__ o16) {
  __shared__ __align__(16) _Float16 P[8][16][64];
  const int wave = threadIdx.x >> 5, lane = threadIdx.x & 31;
  const int head = wave;
  const int t0 = blockIdx.x * 16;
  const size_t rowbase = (size_t)blockIdx.y * T_DIM;
  const int m = lane & 15, colg = lane >> 4, kb = colg * 16, ncol = lane & 15;

  // zero this wave's P region (cols 48..63 stay zero = K padding)
  {
    v8h z;
    for (int i = 0; i < 8; ++i) z[i] = (_Float16)0.f;
    _Float16* pw = &P[wave][0][0] + lane * 32;
    for (int i = 0; i < 4; ++i) *(v8h*)(pw + i * 8) = z;
  }

  // Q fragment (A layout), already scaled by 1/sqrt(DK)
  v16h aq;
  {
    const _Float16* qp = q16 + (rowbase + t0 + m) * 256 + head * 32 + kb;
    v8h lo = *(const v8h*)qp, hi = *(const v8h*)(qp + 8);
    for (int i = 0; i < 8; ++i) { aq[i] = lo[i]; aq[8 + i] = hi[i]; }
  }

  // Scores: 3 key tiles covering [t0-32, t0+15]
  v8f S[3];
  for (int c = 0; c < 3; ++c) {
    int jb = t0 - 32 + c * 16;
    int j = jb + ncol;
    v16h bk;
    if (j >= 0) {
      const _Float16* kp = k16 + (rowbase + j) * 256 + head * 32 + colg * 16;
      v8h lo = *(const v8h*)kp, hi = *(const v8h*)(kp + 8);
      for (int i = 0; i < 8; ++i) { bk[i] = lo[i]; bk[8 + i] = hi[i]; }
    } else {
      for (int i = 0; i < 16; ++i) bk[i] = (_Float16)0.f;
    }
    v8f z = {};
    S[c] = __builtin_amdgcn_wmma_f32_16x16x32_f16(false, aq, false, bk,
                                                  (short)0, z, false, false);
  }

  // Masked softmax, row-wise (rows live in vgpr index + lane half)
  for (int r = 0; r < 8; ++r) {
    int mr = r + 8 * colg;
    int t = t0 + mr;
    float s[3];
    for (int c = 0; c < 3; ++c) {
      int j = t0 - 32 + c * 16 + ncol;
      bool ok = (j >= 0) && (j <= t) && (j > t - 32);
      s[c] = ok ? S[c][r] : -3.0e38f;
    }
    float mx = fmaxf(s[0], fmaxf(s[1], s[2]));
    for (int off = 8; off >= 1; off >>= 1) mx = fmaxf(mx, __shfl_xor(mx, off, 32));
    float e[3], sm = 0.f;
    for (int c = 0; c < 3; ++c) { e[c] = __expf(s[c] - mx); sm += e[c]; }
    for (int off = 8; off >= 1; off >>= 1) sm += __shfl_xor(sm, off, 32);
    float inv = 1.0f / sm;
    for (int c = 0; c < 3; ++c) P[wave][mr][c * 16 + ncol] = (_Float16)(e[c] * inv);
  }
  __syncthreads();

  // P fragments (A layout), K padded to 64
  v16h ap[2];
  for (int kc = 0; kc < 2; ++kc) {
    const _Float16* pp = &P[wave][m][kc * 32 + kb];
    v8h lo = *(const v8h*)pp, hi = *(const v8h*)(pp + 8);
    for (int i = 0; i < 8; ++i) { ap[kc][i] = lo[i]; ap[kc][8 + i] = hi[i]; }
  }
  // O = P @ V, 2 N-tiles (dk halves) x 2 K-chunks
  for (int nt = 0; nt < 2; ++nt) {
    int dk = nt * 16 + ncol;
    v8f acc = {};
    for (int kc = 0; kc < 2; ++kc) {
      v16h bv;
      for (int jp = 0; jp < 16; ++jp) {
        int kidx = kc * 32 + colg * 16 + jp;
        int j = t0 - 32 + kidx;
        bv[jp] = (j >= 0 && j < T_DIM)
                     ? v16[(rowbase + j) * 256 + head * 32 + dk]
                     : (_Float16)0.f;
      }
      acc = __builtin_amdgcn_wmma_f32_16x16x32_f16(false, ap[kc], false, bv,
                                                   (short)0, acc, false, false);
    }
    for (int r = 0; r < 8; ++r) {
      int t = t0 + r + 8 * colg;
      o16[(rowbase + t) * 256 + head * 32 + dk] = (_Float16)acc[r];
    }
  }
}

// ---------------------------------------------------------------------------
// Output projection with residual + bias: h += o16 @ wo + bo (f32 in place)
// ---------------------------------------------------------------------------
__global__ void __launch_bounds__(256) outproj_kernel(
    const _Float16* __restrict__ A, const _Float16* __restrict__ woT,
    const float* __restrict__ bo, float* __restrict__ h) {
  __shared__ __align__(16) _Float16 As[32][256];
  const int tid = threadIdx.x;
  const size_t r0 = (size_t)blockIdx.x * 32;

  if (tid < 32) tdm_load_tile_32x256(A + r0 * 256, &As[0][0]);
  __syncthreads();

  const int wave = tid >> 5, lane = tid & 31;
  const int mt = wave >> 2, nt0 = (wave & 3) * 4;
  const int m = lane & 15, colg = lane >> 4, kb = colg * 16, ncol = lane & 15;

  v8f acc[4] = {{}, {}, {}, {}};
  for (int ks = 0; ks < 8; ++ks) {
    v16h a;
    const _Float16* ap = &As[mt * 16 + m][ks * 32 + kb];
    v8h lo = *(const v8h*)ap, hi = *(const v8h*)(ap + 8);
    for (int i = 0; i < 8; ++i) { a[i] = lo[i]; a[8 + i] = hi[i]; }
    for (int i = 0; i < 4; ++i) {
      int n = (nt0 + i) * 16 + ncol;
      const _Float16* bp = woT + (size_t)n * 256 + ks * 32 + colg * 16;
      __builtin_prefetch(bp + 32, 0, 1);
      v8h blo = *(const v8h*)bp, bhi = *(const v8h*)(bp + 8);
      v16h bf;
      for (int j = 0; j < 8; ++j) { bf[j] = blo[j]; bf[8 + j] = bhi[j]; }
      acc[i] = __builtin_amdgcn_wmma_f32_16x16x32_f16(false, a, false, bf,
                                                      (short)0, acc[i], false, false);
    }
  }
  for (int i = 0; i < 4; ++i) {
    int n = (nt0 + i) * 16 + ncol;
    float bias = bo[n];
    for (int r = 0; r < 8; ++r) {
      size_t row = r0 + mt * 16 + r + 8 * colg;
      h[row * 256 + n] = h[row * 256 + n] + acc[i][r] + bias;
    }
  }
}

// ---------------------------------------------------------------------------
// Mean over T -> [B, D] f32
// ---------------------------------------------------------------------------
__global__ void __launch_bounds__(256) mean_kernel(const float* __restrict__ h,
                                                   float* __restrict__ out) {
  int idx = blockIdx.x * 256 + threadIdx.x;  // 0..1023
  int b = idx >> 8, d = idx & 255;
  const float* p = h + (size_t)b * T_DIM * 256 + d;
  float s = 0.f;
  for (int t = 0; t < T_DIM; ++t) s += p[t * 256];
  out[idx] = s * (1.f / T_DIM);
}

// ---------------------------------------------------------------------------
extern "C" void kernel_launch(void* const* d_in, const int* in_sizes, int n_in,
                              void* d_out, int out_size, void* d_ws,
                              size_t ws_size, hipStream_t stream) {
  (void)in_sizes; (void)n_in; (void)out_size; (void)ws_size;
  const float* x      = (const float*)d_in[0];
  const float* s1_g   = (const float*)d_in[1];
  const float* s1_b   = (const float*)d_in[2];
  const float* fc1_w  = (const float*)d_in[3];
  const float* fc1_b  = (const float*)d_in[4];
  const float* fc2_w  = (const float*)d_in[5];
  const float* fc2_b  = (const float*)d_in[6];
  const float* proj_w = (const float*)d_in[7];
  const float* proj_b = (const float*)d_in[8];
  const float* ln_g   = (const float*)d_in[9];
  const float* ln_b   = (const float*)d_in[10];
  const float* wq     = (const float*)d_in[11];
  const float* wk     = (const float*)d_in[12];
  const float* wv     = (const float*)d_in[13];
  const float* wo     = (const float*)d_in[14];
  const float* bo     = (const float*)d_in[15];

  char* w = (char*)d_ws;
  float*    h    = (float*)(w);                                  // 8 MB
  _Float16* n16  = (_Float16*)(w + (size_t)(8u << 20));          // 4 MB
  _Float16* q16  = (_Float16*)(w + (size_t)(12u << 20));
  _Float16* k16  = (_Float16*)(w + (size_t)(16u << 20));
  _Float16* v16p = (_Float16*)(w + (size_t)(20u << 20));
  _Float16* o16  = (_Float16*)(w + (size_t)(24u << 20));
  _Float16* fc2T = (_Float16*)(w + (size_t)(28u << 20));         // 256 KB
  _Float16* wT   = (_Float16*)(w + (size_t)(28u << 20) + 512 * 256 * 2);

  // weights -> transposed f16 (once per call; all L2-resident afterwards)
  convw_kernel<<<(512 * 256 + 255) / 256, 256, 0, stream>>>(fc2_w, fc2T, 512, 256);
  for (int l = 0; l < 2; ++l) {
    const float* srcs[4] = {wq + l * 65536, wk + l * 65536,
                            wv + l * 65536, wo + l * 65536};
    for (int j = 0; j < 4; ++j)
      convw_kernel<<<256, 256, 0, stream>>>(srcs[j], wT + (size_t)(l * 4 + j) * 65536,
                                            256, 256);
  }

  stage1_kernel<<<BT / 16, 128, 0, stream>>>(x, s1_g, s1_b, fc1_w, fc1_b, fc2T,
                                             fc2_b, proj_w, proj_b, h);

  for (int l = 0; l < 2; ++l) {
    const _Float16* wqT = wT + (size_t)(l * 4 + 0) * 65536;
    const _Float16* wkT = wT + (size_t)(l * 4 + 1) * 65536;
    const _Float16* wvT = wT + (size_t)(l * 4 + 2) * 65536;
    const _Float16* woT = wT + (size_t)(l * 4 + 3) * 65536;
    ln_kernel<<<BT / 8, 256, 0, stream>>>(h, ln_g + l * 256, ln_b + l * 256, n16);
    qkv_kernel<<<dim3(BT / 32, 3), 256, 0, stream>>>(n16, wqT, wkT, wvT, q16, k16,
                                                     v16p);
    attn_kernel<<<dim3(T_DIM / 16, B_DIM), 256, 0, stream>>>(q16, k16, v16p, o16);
    outproj_kernel<<<BT / 32, 256, 0, stream>>>(o16, woT, bo + l * 256, h);
  }

  mean_kernel<<<4, 256, 0, stream>>>(h, (float*)d_out);
}
